// SafariBackbone_8624294330617
// MI455X (gfx1250) — matrix operand order repaired and checked by reading, not compiled, not run on hardware
//
#include <hip/hip_runtime.h>
#include <math.h>

// Sizes from the reference
#define Bx   128
#define Tx   128
#define NF   64
#define Hh   128
#define G3H  384   // 3*H
#define Kk   8
#define KMIT 25

typedef __attribute__((ext_vector_type(16))) _Float16     v16h;
typedef __attribute__((ext_vector_type(8)))  float        v8f;
typedef __attribute__((ext_vector_type(4)))  unsigned int v4u;

union Frag16 {
    v16h v;
    _Float16 h[16];
    v4u  q[2];      // two 128-bit halves -> ds_load_b128
};

// Branch-free fast activations: v_exp_f32 + v_rcp_f32 only (no IEEE div
// sequence, no libm tanh magnitude-branch -> no EXEC churn in the WMMA loop).
__device__ __forceinline__ float fast_sigmoid(float xv) {
    return __builtin_amdgcn_rcpf(1.0f + __expf(-xv));
}
__device__ __forceinline__ float fast_tanh(float xv) {
    return 2.0f * __builtin_amdgcn_rcpf(1.0f + __expf(-2.0f * xv)) - 1.0f;
}

// ---------------------------------------------------------------------------
// GRU scan: one workgroup per feature f. 8 waves; wave w owns gate-column
// group j=w (columns 16j..16j+15 of each gate) and sweeps the 8 batch tiles.
// B fragments (whh) are loop-invariant -> loaded once into 96 VGPRs and held
// for all 128 time steps. A fragments are two ds_load_b128 per k-step.
//
// LDS layout (dynamic):
//   h16  : [2][128][128] f16 (double buffer)    65536 B
//   h32  : [128][128] f32                       65536 B
//   xt   : [128] f32                              512 B
//   lens : [128] i32                              512 B
// total = 132096 B
// ---------------------------------------------------------------------------
#define GRU_LDS_BYTES 132096

__global__ __launch_bounds__(256) void gru_scan_kernel(
    const float* __restrict__ x,       // [B,T,NF]
    const int*   __restrict__ lengths, // [B]
    const float* __restrict__ wih,     // [NF,3H]
    const float* __restrict__ whh,     // [NF,3H,H]
    const float* __restrict__ bih,     // [NF,3H]
    const float* __restrict__ bhh,     // [NF,3H]
    float*       __restrict__ tokens)  // [B,NF,H]
{
    extern __shared__ char gru_smem[];
    _Float16* h16 = (_Float16*)gru_smem;                 // 65536 B
    float*    h32 = (float*)(gru_smem + 65536);          // 65536 B
    float*    xt  = (float*)(gru_smem + 131072);         // 512 B
    int*      lens = (int*)(gru_smem + 131584);          // 512 B

    const int f     = blockIdx.x;
    const int tid   = threadIdx.x;
    const int lane  = tid & 31;
    const int wid   = tid >> 5;               // 0..7
    const int l15   = lane & 15;
    const int hiGrp = (lane >= 16) ? 1 : 0;

    const int j    = wid;                     // owned gate-column group
    const int c    = 16 * j + l15;            // gate column 0..127 (fixed/lane)

    // ---- loop-invariant B fragments: B[k,n] = whh[g(n), k], gathered from
    // global (one pass through L2), converted to f16, resident in VGPRs.
    const float* whhf = whh + (size_t)f * G3H * Hh;
    Frag16 Br[4], Bz[4], Bn[4];
    #pragma unroll
    for (int kb = 0; kb < 4; ++kb) {
        const int kbase = 32 * kb + (hiGrp ? 8 : 0);
        #pragma unroll
        for (int e = 0; e < 16; ++e) {
            const int kk = kbase + (e & 7) + ((e & 8) ? 16 : 0);
            Br[kb].h[e] = (_Float16)whhf[(0   + c) * Hh + kk];
            Bz[kb].h[e] = (_Float16)whhf[(128 + c) * Hh + kk];
            Bn[kb].h[e] = (_Float16)whhf[(256 + c) * Hh + kk];
        }
    }
    // ---- loop-invariant gate scalars for this lane's column
    // r/z gates always see (bih + bhh) summed -> presum.
    const float wr  = wih[f * G3H +       c];
    const float wz  = wih[f * G3H + 128 + c];
    const float wn  = wih[f * G3H + 256 + c];
    const float sbr = bih[f * G3H +       c] + bhh[f * G3H +       c];
    const float sbz = bih[f * G3H + 128 + c] + bhh[f * G3H + 128 + c];
    const float bn_ = bih[f * G3H + 256 + c];
    const float hn_ = bhh[f * G3H + 256 + c];

    for (int i = tid; i < Bx; i += 256) lens[i] = lengths[i];
    for (int i = tid; i < Bx * Hh; i += 256) {
        h32[i] = 0.0f;
        h16[i] = (_Float16)0.0f;              // buffer 0 (t=0 reads this)
    }
    __syncthreads();

    for (int t = 0; t < Tx; ++t) {
        const int p = t & 1;
        _Float16* hcur = h16 + p * (Bx * Hh);
        _Float16* hnxt = h16 + (p ^ 1) * (Bx * Hh);

        if (tid < Bx) {
            xt[tid] = x[(size_t)tid * (Tx * NF) + (size_t)t * NF + f];
            if (t + 1 < Tx)   // gfx1250 global_prefetch_b8
                __builtin_prefetch(&x[(size_t)tid * (Tx * NF) + (size_t)(t + 1) * NF + f], 0, 1);
        }
        __syncthreads();

        for (int m = 0; m < 8; ++m) {
            v8f accr = {}; v8f accz = {}; v8f accn = {};
            const int mrow = 16 * m + l15;

            #pragma unroll
            for (int kb = 0; kb < 4; ++kb) {
                // A fragment: rows 16m.., cols 32kb.. ; VGPR0-3 = 8 contiguous
                // K at (32kb + 8*hiGrp), VGPR4-7 = 8 contiguous K at +16.
                Frag16 a;
                const v4u* pa = (const v4u*)(hcur + mrow * Hh + 32 * kb + (hiGrp ? 8 : 0));
                a.q[0] = pa[0];
                a.q[1] = pa[2];
                accr = __builtin_amdgcn_wmma_f32_16x16x32_f16(false, a.v, false, Br[kb].v, (short)0, accr, false, false);
                accz = __builtin_amdgcn_wmma_f32_16x16x32_f16(false, a.v, false, Bz[kb].v, (short)0, accz, false, false);
                accn = __builtin_amdgcn_wmma_f32_16x16x32_f16(false, a.v, false, Bn[kb].v, (short)0, accn, false, false);
            }

            // ---- gate update in registers (C/D layout: VGPR i, lane -> element)
            #pragma unroll
            for (int i = 0; i < 8; ++i) {
                const int bb = 16 * m + i + (hiGrp ? 8 : 0);
                const float xv = xt[bb];
                const float r  = fast_sigmoid(xv * wr + sbr + accr[i]);
                const float z  = fast_sigmoid(xv * wz + sbz + accz[i]);
                const float nn = fast_tanh(xv * wn + bn_ + r * (accn[i] + hn_));
                const float hold = h32[bb * Hh + c];
                const float hnew = (1.0f - z) * nn + z * hold;
                h32[bb * Hh + c]  = hnew;
                hnxt[bb * Hh + c] = (_Float16)hnew;
                if (lens[bb] == t + 1)
                    tokens[(size_t)bb * (NF * Hh) + (size_t)f * Hh + c] = hnew;
            }
        }
        __syncthreads();   // protect xt / hnxt before next step
    }
}

// ---------------------------------------------------------------------------
// node[f,h] = mean_b tokens[b,f,h]
// ---------------------------------------------------------------------------
__global__ __launch_bounds__(128) void node_mean_kernel(
    const float* __restrict__ tokens, float* __restrict__ node)
{
    const int f = blockIdx.x, h = threadIdx.x;
    float s = 0.0f;
    for (int b = 0; b < Bx; ++b) s += tokens[(size_t)b * (NF * Hh) + (size_t)f * Hh + h];
    node[f * Hh + h] = s * (1.0f / (float)Bx);
}

// ---------------------------------------------------------------------------
// Lloyd k-means (K=8, 25 iters, deterministic init) + normalized adjacency.
// Single block.
// ---------------------------------------------------------------------------
__global__ __launch_bounds__(256) void kmeans_adj_kernel(
    const float* __restrict__ node_g, float* __restrict__ A_out)
{
    __shared__ float nodes[NF * Hh];     // 32 KB
    __shared__ float centers[Kk * Hh];   // 4 KB
    __shared__ float sums[Kk * Hh];      // 4 KB
    __shared__ float adj[NF * NF];       // 16 KB
    __shared__ int   labels[NF];
    __shared__ int   counts[Kk];
    __shared__ float norms[NF];
    __shared__ float dinv[NF];

    const int tid = threadIdx.x;
    for (int i = tid; i < NF * Hh; i += 256) nodes[i] = node_g[i];
    __syncthreads();
    for (int i = tid; i < Kk * Hh; i += 256)
        centers[i] = nodes[(i / Hh) * 8 * Hh + (i % Hh)];   // centers k = node[8k]
    __syncthreads();

    for (int it = 0; it <= KMIT; ++it) {          // last pass = final assignment
        if (tid < NF) {
            float best = 3.4e38f; int bl = 0;
            for (int k = 0; k < Kk; ++k) {
                float d = 0.0f;
                for (int h = 0; h < Hh; ++h) {
                    const float df = nodes[tid * Hh + h] - centers[k * Hh + h];
                    d += df * df;
                }
                if (d < best) { best = d; bl = k; }
            }
            labels[tid] = bl;
        }
        __syncthreads();
        if (it == KMIT) break;
        for (int i = tid; i < Kk * Hh; i += 256) sums[i] = 0.0f;
        __syncthreads();
        if (tid < Hh) {
            for (int n = 0; n < NF; ++n)
                sums[labels[n] * Hh + tid] += nodes[n * Hh + tid];
        } else if (tid < Hh + Kk) {
            const int k = tid - Hh; int cnt = 0;
            for (int n = 0; n < NF; ++n) cnt += (labels[n] == k) ? 1 : 0;
            counts[k] = cnt;
        }
        __syncthreads();
        for (int i = tid; i < Kk * Hh; i += 256) {
            const int k = i / Hh;
            if (counts[k] > 0) centers[i] = sums[i] / (float)counts[k];
        }
        __syncthreads();
    }

    if (tid < NF) {
        float s = 0.0f;
        for (int h = 0; h < Hh; ++h) s += nodes[tid * Hh + h] * nodes[tid * Hh + h];
        norms[tid] = fmaxf(sqrtf(s), 1e-12f);
    }
    __syncthreads();
    for (int i = tid; i < NF * NF; i += 256) {
        const int n = i >> 6, mm = i & 63;
        float v = 0.0f;
        if (labels[n] == labels[mm]) {
            float dot = 0.0f;
            for (int h = 0; h < Hh; ++h) dot += nodes[n * Hh + h] * nodes[mm * Hh + h];
            v = fmaxf(dot / (norms[n] * norms[mm]), 0.0f);
        }
        adj[i] = v;
    }
    __syncthreads();
    if (tid < NF) {
        float d = 0.0f;
        for (int mm = 0; mm < NF; ++mm) d += adj[tid * NF + mm];
        dinv[tid] = (d > 0.0f) ? rsqrtf(d) : 0.0f;
    }
    __syncthreads();
    for (int i = tid; i < NF * NF; i += 256) {
        const int n = i >> 6, mm = i & 63;
        A_out[i] = dinv[n] * adj[i] * dinv[mm];
    }
}

// ---------------------------------------------------------------------------
// GCN layer: out[b] = relu( (A @ in[b]) @ w + bias )
// ---------------------------------------------------------------------------
__global__ __launch_bounds__(256) void gcn_layer_kernel(
    const float* __restrict__ A, const float* __restrict__ in,
    const float* __restrict__ w, const float* __restrict__ bias,
    float* __restrict__ out)
{
    extern __shared__ float gcn_smem[];
    float* tin = gcn_smem;            // [64][128]
    float* t1  = gcn_smem + NF * Hh;  // [64][128]
    const int b = blockIdx.x, tid = threadIdx.x;
    const float* inb = in + (size_t)b * (NF * Hh);
    for (int i = tid; i < NF * Hh; i += 256) tin[i] = inb[i];
    __syncthreads();
    for (int i = tid; i < NF * Hh; i += 256) {
        const int n = i >> 7, h = i & 127;
        float s = 0.0f;
        for (int mm = 0; mm < NF; ++mm) s += A[n * NF + mm] * tin[mm * Hh + h];
        t1[i] = s;
    }
    __syncthreads();
    float* ob = out + (size_t)b * (NF * Hh);
    for (int i = tid; i < NF * Hh; i += 256) {
        const int n = i >> 7, hp = i & 127;
        float s = bias[hp];
        for (int h = 0; h < Hh; ++h) s += t1[n * Hh + h] * w[h * Hh + hp];
        ob[i] = fmaxf(s, 0.0f);
    }
}

// ---------------------------------------------------------------------------
// Fused single-query attention + output projection (per batch element)
// ---------------------------------------------------------------------------
__global__ __launch_bounds__(128) void attention_kernel(
    const float* __restrict__ g,
    const float* __restrict__ wq, const float* __restrict__ bq,
    const float* __restrict__ wk, const float* __restrict__ bk,
    const float* __restrict__ wv, const float* __restrict__ bv,
    const float* __restrict__ wo, const float* __restrict__ bo,
    float* __restrict__ out)
{
    __shared__ float gs[NF * Hh];   // 32 KB
    __shared__ float qs[Hh];
    __shared__ float score[NF];
    __shared__ float wsm[NF];
    __shared__ float ctx[Hh];

    const int b = blockIdx.x, tid = threadIdx.x;
    const float* gb = g + (size_t)b * (NF * Hh);
    for (int i = tid; i < NF * Hh; i += 128) gs[i] = gb[i];
    __syncthreads();

    {   // q = g[:, -1, :] @ wq + bq
        float s = bq[tid];
        for (int mm = 0; mm < Hh; ++mm) s += gs[63 * Hh + mm] * wq[mm * Hh + tid];
        qs[tid] = s;
    }
    __syncthreads();

    if (tid < NF) {  // score[n] = (g[n] @ wk + bk) . q / sqrt(H)
        float s = 0.0f;
        for (int h = 0; h < Hh; ++h) {
            float kv = bk[h];
            for (int mm = 0; mm < Hh; ++mm) kv += gs[tid * Hh + mm] * wk[mm * Hh + h];
            s += kv * qs[h];
        }
        score[tid] = s * 0.08838834764831845f;   // 1/sqrt(128)
    }
    __syncthreads();

    float mx = -3.4e38f;
    for (int n = 0; n < NF; ++n) mx = fmaxf(mx, score[n]);
    float ssum = 0.0f;
    for (int n = 0; n < NF; ++n) ssum += __expf(score[n] - mx);
    if (tid < NF) wsm[tid] = __expf(score[tid] - mx) / ssum;
    __syncthreads();

    {   // ctx[h] = sum_n w[n] * (g[n] @ wv + bv)[h]
        float s = 0.0f;
        for (int n = 0; n < NF; ++n) {
            float vv = bv[tid];
            for (int mm = 0; mm < Hh; ++mm) vv += gs[n * Hh + mm] * wv[mm * Hh + tid];
            s += wsm[n] * vv;
        }
        ctx[tid] = s;
    }
    __syncthreads();

    {   // out = relu(ctx @ wo + bo)
        float s = bo[tid];
        for (int h = 0; h < Hh; ++h) s += ctx[h] * wo[h * Hh + tid];
        out[(size_t)b * Hh + tid] = fmaxf(s, 0.0f);
    }
}

// ---------------------------------------------------------------------------
extern "C" void kernel_launch(void* const* d_in, const int* in_sizes, int n_in,
                              void* d_out, int out_size, void* d_ws, size_t ws_size,
                              hipStream_t stream)
{
    const float* x        = (const float*)d_in[0];
    const int*   lengths  = (const int*)  d_in[1];
    const float* gru_wih  = (const float*)d_in[2];
    const float* gru_whh  = (const float*)d_in[3];
    const float* gru_bih  = (const float*)d_in[4];
    const float* gru_bhh  = (const float*)d_in[5];
    const float* w1 = (const float*)d_in[6];
    const float* b1 = (const float*)d_in[7];
    const float* w2 = (const float*)d_in[8];
    const float* b2 = (const float*)d_in[9];
    const float* wq = (const float*)d_in[10];
    const float* bq = (const float*)d_in[11];
    const float* wk = (const float*)d_in[12];
    const float* bk = (const float*)d_in[13];
    const float* wv = (const float*)d_in[14];
    const float* bv = (const float*)d_in[15];
    const float* wo = (const float*)d_in[16];
    const float* bo = (const float*)d_in[17];
    float* out = (float*)d_out;

    // Workspace carve-up (f32): tokens | node | A | g1 | g2  (~12.1 MB)
    float* tokens = (float*)d_ws;
    float* node   = tokens + (size_t)Bx * NF * Hh;   // +1,048,576
    float* Amat   = node   + (size_t)NF * Hh;        // +8,192
    float* g1     = Amat   + (size_t)NF * NF;        // +4,096
    float* g2     = g1     + (size_t)Bx * NF * Hh;   // +1,048,576

    gru_scan_kernel<<<NF, 256, GRU_LDS_BYTES, stream>>>(
        x, lengths, gru_wih, gru_whh, gru_bih, gru_bhh, tokens);
    node_mean_kernel<<<NF, Hh, 0, stream>>>(tokens, node);
    kmeans_adj_kernel<<<1, 256, 0, stream>>>(node, Amat);
    gcn_layer_kernel<<<Bx, 256, 2 * NF * Hh * sizeof(float), stream>>>(Amat, tokens, w1, b1, g1);
    gcn_layer_kernel<<<Bx, 256, 2 * NF * Hh * sizeof(float), stream>>>(Amat, g1, w2, b2, g2);
    attention_kernel<<<Bx, Hh, 0, stream>>>(g2, wq, bq, wk, bk, wv, bv, wo, bo, out);
}
